// PAM_13932873908716
// MI455X (gfx1250) — compile-verified, matrix operand order checked by reading, not run on hardware
//
#include <hip/hip_runtime.h>

// ---------------------------------------------------------------------------
// PAM (Parallax Attention Module) for MI455X / gfx1250, wave32 + WMMA f16.
// All GEMM-shaped work (3x3 convs via im2col, 1x1 Q/K convs, QtK attention
// scores, M @ x warps) runs through one wave-tiled, register double-buffered
// V_WMMA_F32_16X16X32_F16 kernel; elementwise stages are coalesced VALU.
// ---------------------------------------------------------------------------

typedef __attribute__((ext_vector_type(16))) _Float16 v16h;
typedef __attribute__((ext_vector_type(8)))  float    v8f;

#define BB  4
#define CC  64
#define CH  32
#define HH  128
#define WW  256
#define HW  (HH * WW)          // 32768
#define BH  (BB * HH)          // 512
#define EPS 1e-5f

union U32H2 { unsigned int u; _Float16 h[2]; };

// ---------------------------------------------------------------------------
// Generic WMMA GEMM:  D[M x N] (f32) = A[M x K] (f16, row-major, lda=K)
//                                    x Bt[N x K]^T (f16, row-major, ldb=K)
//                                    (+ D preload when ADDC)
// One wave per (16*MT) x (16*NT) tile; ping-pong double-buffered fragments so
// next-chunk global_load_b128s overlap the current WMMA group. Fragment
// addresses are running pointers (+64B/chunk). waves_per_eu(2) keeps the
// VGPR budget high enough to avoid scratch spills.
// (M/16)%MT==0, (N/16)%NT==0, K%32==0, K>=64 required.
// ---------------------------------------------------------------------------
template <int MT, int NT, bool ADDC>
__global__ __launch_bounds__(256)
__attribute__((amdgpu_waves_per_eu(2)))
void k_gemm(const _Float16* __restrict__ A,
            const _Float16* __restrict__ Bt,
            float* __restrict__ C,
            int M, int N, int K,
            long long sA, long long sB, long long sC)
{
  const int lane = threadIdx.x & 31;
  const int wave = threadIdx.x >> 5;
  const int l16  = lane & 15;
  const int g    = lane >> 4;

  const int tilesM = (M >> 4) / MT;
  const int tilesN = (N >> 4) / NT;
  const int tile   = blockIdx.x * (blockDim.x >> 5) + wave;
  if (tile >= tilesM * tilesN) return;           // wave-uniform exit
  const int mtT = (tile % tilesM) * MT;
  const int ntT = (tile / tilesM) * NT;

  const long long bz = blockIdx.z;
  A  += bz * sA;
  Bt += bz * sB;
  C  += bz * sC;

  // K offsets of the 8 dword slots of a 16x32 f16 fragment for this lane.
  int kof[8];
#pragma unroll
  for (int j = 0; j < 8; ++j)
    kof[j] = ((j < 4) ? 0 : 16) + g * 8 + (j & 3) * 2;

  v8f acc[MT][NT];
#pragma unroll
  for (int m = 0; m < MT; ++m)
#pragma unroll
    for (int n = 0; n < NT; ++n) {
      if (ADDC) {
#pragma unroll
        for (int v = 0; v < 8; ++v)
          acc[m][n][v] =
              C[(long long)((mtT + m) * 16 + v + 8 * g) * N + (ntT + n) * 16 + l16];
      } else {
#pragma unroll
        for (int v = 0; v < 8; ++v) acc[m][n][v] = 0.0f;
      }
    }

  // running per-fragment pointers, advanced one 32-K chunk per load
  const _Float16* aPtr[MT];
  const _Float16* bPtr[NT];
#pragma unroll
  for (int m = 0; m < MT; ++m)
    aPtr[m] = A + (long long)((mtT + m) * 16 + l16) * K;
#pragma unroll
  for (int n = 0; n < NT; ++n)
    bPtr[n] = Bt + (long long)((ntT + n) * 16 + l16) * K;

  auto ldA = [&](v16h* dst) {
#pragma unroll
    for (int m = 0; m < MT; ++m) {
#pragma unroll
      for (int j = 0; j < 8; ++j) {
        U32H2 u; u.u = *(const unsigned int*)(aPtr[m] + kof[j]);
        dst[m][2 * j] = u.h[0]; dst[m][2 * j + 1] = u.h[1];
      }
      aPtr[m] += 32;
    }
  };
  auto ldB = [&](v16h* dst) {
#pragma unroll
    for (int n = 0; n < NT; ++n) {
#pragma unroll
      for (int j = 0; j < 8; ++j) {
        U32H2 u; u.u = *(const unsigned int*)(bPtr[n] + kof[j]);
        dst[n][2 * j] = u.h[0]; dst[n][2 * j + 1] = u.h[1];
      }
      bPtr[n] += 32;
    }
  };
  auto mma = [&](v16h* a, v16h* b) {
#pragma unroll
    for (int m = 0; m < MT; ++m)
#pragma unroll
      for (int n = 0; n < NT; ++n)
        acc[m][n] = __builtin_amdgcn_wmma_f32_16x16x32_f16(
            false, a[m], false, b[n], (short)0, acc[m][n], false, false);
  };

  // ping-pong double-buffered K loop
  const int steps = K >> 5;
  v16h aP[MT], bP[NT], aQ[MT], bQ[NT];
  ldA(aP); ldB(bP);
  int s = 0;
  while (true) {
    if (s + 1 < steps) { ldA(aQ); ldB(bQ); }
    mma(aP, bP);
    ++s;
    if (s >= steps) break;
    if (s + 1 < steps) { ldA(aP); ldB(bP); }
    mma(aQ, bQ);
    ++s;
    if (s >= steps) break;
  }

#pragma unroll
  for (int m = 0; m < MT; ++m)
#pragma unroll
    for (int n = 0; n < NT; ++n)
#pragma unroll
      for (int v = 0; v < 8; ++v)
        C[(long long)((mtT + m) * 16 + v + 8 * g) * N + (ntT + n) * 16 + l16] =
            acc[m][n][v];
}

// ---------------------------------------------------------------------------
// Elementwise / packing kernels
// ---------------------------------------------------------------------------
__global__ void k_f32_to_f16(const float* __restrict__ s, _Float16* __restrict__ d, int n)
{
  for (int i = blockIdx.x * 256 + threadIdx.x; i < n; i += gridDim.x * 256)
    d[i] = (_Float16)s[i];
}

__global__ void k_add(float* __restrict__ d, const float* __restrict__ s, int n)
{
  for (int i = blockIdx.x * 256 + threadIdx.x; i < n; i += gridDim.x * 256)
    d[i] += s[i];
}

// im2col: x (Cin,H,W) f32 -> col (HW x Cin*9) f16, k = c*9 + r*3 + s (pad=1)
__global__ void k_im2col(const float* __restrict__ x, _Float16* __restrict__ col,
                         int Cin, int n)
{
  const int K = Cin * 9;
  for (int idx = blockIdx.x * 256 + threadIdx.x; idx < n; idx += gridDim.x * 256) {
    int k   = idx % K;
    int pix = idx / K;
    int c  = k / 9;
    int rs = k - c * 9;
    int r  = rs / 3, s = rs - r * 3;
    int h  = pix >> 8;           // W = 256
    int w  = pix & 255;
    int hh = h + r - 1, ww = w + s - 1;
    float v = 0.0f;
    if (hh >= 0 && hh < HH && ww >= 0 && ww < WW)
      v = x[c * HW + hh * WW + ww];
    col[idx] = (_Float16)v;
  }
}

// bias + BN + PReLU on GEMM output (HW x Cout), writes CHW f32 (+ pixel-major f16)
__global__ void k_bn_prelu(const float* __restrict__ gin,
                           float* __restrict__ out_chw,
                           _Float16* __restrict__ out_pix,   // may be nullptr
                           const float* __restrict__ bias,
                           const float* __restrict__ gg, const float* __restrict__ bb,
                           const float* __restrict__ mm, const float* __restrict__ vv,
                           const float* __restrict__ alpha,
                           int Cout, int n)
{
  const float a = alpha[0];
  for (int idx = blockIdx.x * 256 + threadIdx.x; idx < n; idx += gridDim.x * 256) {
    int c  = idx / HW;
    int hw = idx - c * HW;
    float val = gin[(long long)hw * Cout + c] + bias[c];
    val = gg[c] * (val - mm[c]) * __frsqrt_rn(vv[c] + EPS) + bb[c];
    val = (val >= 0.0f) ? val : a * val;
    out_chw[idx] = val;
    if (out_pix) out_pix[(long long)hw * Cout + c] = (_Float16)val;
  }
}

// Q/K post: subtract per-(b,h,c) mean over W (bias cancels), pack (BH, W, C) f16
__global__ void k_qk_post(const float* __restrict__ raw, _Float16* __restrict__ P)
{
  int t = blockIdx.x * 256 + threadIdx.x;       // BH * C = 32768
  if (t >= BH * CC) return;
  int c  = t & (CC - 1);
  int bh = t >> 6;
  const float* src = raw + (long long)bh * WW * CC + c;
  float s = 0.0f;
  for (int w = 0; w < WW; ++w) s += src[w * CC];
  float mean = s * (1.0f / WW);
  _Float16* dst = P + (long long)bh * WW * CC + c;
  for (int w = 0; w < WW; ++w) dst[w * CC] = (_Float16)(src[w * CC] - mean);
}

// row softmax over 256 + fused disparity; one block (256 thr) per row
__global__ void k_softmax_disp(const float* __restrict__ X,
                               float* __restrict__ Mout, float* __restrict__ disp)
{
  __shared__ float sm[256];
  const long long r = blockIdx.x;
  const int t = threadIdx.x;
  float v = X[r * WW + t];
  sm[t] = v; __syncthreads();
  for (int s = 128; s > 0; s >>= 1) { if (t < s) sm[t] = fmaxf(sm[t], sm[t + s]); __syncthreads(); }
  float mx = sm[0]; __syncthreads();
  float e = __expf(v - mx);
  sm[t] = e; __syncthreads();
  for (int s = 128; s > 0; s >>= 1) { if (t < s) sm[t] += sm[t + s]; __syncthreads(); }
  float inv = 1.0f / sm[0]; __syncthreads();
  float p = e * inv;
  Mout[r * WW + t] = p;
  sm[t] = p * (float)t; __syncthreads();
  for (int s = 128; s > 0; s >>= 1) { if (t < s) sm[t] += sm[t + s]; __syncthreads(); }
  if (t == 0) disp[r] = sm[0];
}

// V = tanh(5 * sum_v relax(Ma)[p,q,v] * Mb[p,v,q]); one block per (p,q)
__global__ void k_relax_V(const float* __restrict__ Ma, const float* __restrict__ Mb,
                          float* __restrict__ Vout)
{
  __shared__ float sm[256];
  const int p = blockIdx.x >> 8;
  const int q = blockIdx.x & 255;
  const int v = threadIdx.x;
  float rel = 0.0f;
#pragma unroll
  for (int d = -2; d <= 2; ++d) {
    int qq = q + d;
    if (qq >= 0 && qq < WW) rel += Ma[((long long)p * WW + qq) * WW + v];
  }
  float val = rel * Mb[((long long)p * WW + v) * WW + q];
  sm[v] = val; __syncthreads();
  for (int s = 128; s > 0; s >>= 1) { if (v < s) sm[v] += sm[v + s]; __syncthreads(); }
  if (v == 0) Vout[(long long)p * WW + q] = tanhf(5.0f * sm[0]);
}

// x (B,C,H,W) f32 -> xt (BH, C, W) f16  (Bt layout for the warp GEMM)
__global__ void k_xT16(const float* __restrict__ x, _Float16* __restrict__ xt, int n)
{
  for (int idx = blockIdx.x * 256 + threadIdx.x; idx < n; idx += gridDim.x * 256) {
    int w = idx & 255;
    int h = (idx >> 8) & 127;
    int c = (idx >> 15) & 63;
    int b = idx >> 21;
    xt[(((long long)(b * HH + h)) * CC + c) * WW + w] = (_Float16)x[idx];
  }
}

// out = x*(1-V) + xT*V  (xT in (BH, W, C) pixel-major)
__global__ void k_blend(const float* __restrict__ x, const float* __restrict__ xT,
                        const float* __restrict__ V, float* __restrict__ out, int n)
{
  for (int idx = blockIdx.x * 256 + threadIdx.x; idx < n; idx += gridDim.x * 256) {
    int w = idx & 255;
    int h = (idx >> 8) & 127;
    int c = (idx >> 15) & 63;
    int b = idx >> 21;
    long long p = (long long)(b * HH + h);
    float vv = V[p * WW + w];
    out[idx] = x[idx] * (1.0f - vv) + xT[(p * WW + w) * CC + c] * vv;
  }
}

// ---------------------------------------------------------------------------
// Host side
// ---------------------------------------------------------------------------
static void launch_gemm(hipStream_t st, const _Float16* A, const _Float16* Bt, float* C,
                        int M, int N, int K, int batch,
                        long long sA, long long sB, long long sC, bool addC)
{
  const bool wide = (N % 64 == 0);
  const int mt = wide ? 1 : 2;
  const int nt = wide ? 4 : 2;
  const int tiles = ((M / 16) / mt) * ((N / 16) / nt);
  dim3 grid((tiles + 7) / 8, 1, batch);
  dim3 blk(256, 1, 1);
  if (wide) {
    if (addC) k_gemm<1, 4, true ><<<grid, blk, 0, st>>>(A, Bt, C, M, N, K, sA, sB, sC);
    else      k_gemm<1, 4, false><<<grid, blk, 0, st>>>(A, Bt, C, M, N, K, sA, sB, sC);
  } else {
    if (addC) k_gemm<2, 2, true ><<<grid, blk, 0, st>>>(A, Bt, C, M, N, K, sA, sB, sC);
    else      k_gemm<2, 2, false><<<grid, blk, 0, st>>>(A, Bt, C, M, N, K, sA, sB, sC);
  }
}

static inline dim3 gs(long long n) { return dim3((unsigned)((n + 255) / 256), 1, 1); }

// workspace layout (bytes)
static constexpr size_t O_C0   = 0;
static constexpr size_t O_C1   = O_C0   + 134217728ull;  // (BH,W,W) f32
static constexpr size_t O_FEAL = O_C1   + 134217728ull;
static constexpr size_t O_FEAR = O_FEAL + 33554432ull;   // (B,C,H,W) f32
static constexpr size_t O_FL1  = O_FEAR + 33554432ull;
static constexpr size_t O_FR1  = O_FL1  + 33554432ull;
static constexpr size_t O_FL1P = O_FR1  + 33554432ull;   // (B*HW, C) f16
static constexpr size_t O_FR1P = O_FL1P + 16777216ull;
static constexpr size_t O_YB   = O_FR1P + 16777216ull;   // (CH,HW) f32
static constexpr size_t O_G1   = O_YB   + 4194304ull;    // (HW,CH) f32
static constexpr size_t O_G2   = O_G1   + 4194304ull;    // (HW,C)  f32
static constexpr size_t O_COL  = O_G2   + 8388608ull;    // (HW, 576) f16
static constexpr size_t O_QRAW = O_COL  + 37748736ull;   // (B*HW, C) f32
static constexpr size_t O_QP   = O_QRAW + 33554432ull;   // (BH,W,C) f16
static constexpr size_t O_KP   = O_QP   + 16777216ull;
static constexpr size_t O_W1H  = O_KP   + 16777216ull;   // f16 weights
static constexpr size_t O_W2H  = O_W1H  + 147456ull;
static constexpr size_t O_WQH  = O_W2H  + 147456ull;
static constexpr size_t O_WKH  = O_WQH  + 32768ull;
static constexpr size_t O_M16  = O_WKH  + 32768ull;      // (BH,W,W) f16
static constexpr size_t O_XT16 = O_M16  + 67108864ull;   // (BH,C,W) f16
static constexpr size_t O_XT   = O_XT16 + 16777216ull;   // (BH,W,C) f32

// d_out layout (floats)
static constexpr size_t OUT_L  = 0;
static constexpr size_t OUT_R  = 8388608;
static constexpr size_t OUT_D1 = 16777216;
static constexpr size_t OUT_D2 = 16908288;
static constexpr size_t OUT_M0 = 17039360;
static constexpr size_t OUT_M1 = 50593792;
static constexpr size_t OUT_VL = 84148224;
static constexpr size_t OUT_VR = 84279296;

extern "C" void kernel_launch(void* const* d_in, const int* in_sizes, int n_in,
                              void* d_out, int out_size, void* d_ws, size_t ws_size,
                              hipStream_t stream)
{
  (void)in_sizes; (void)n_in; (void)out_size; (void)ws_size;
  char* ws = (char*)d_ws;
  float* out = (float*)d_out;

  const float* x_left  = (const float*)d_in[0];
  const float* x_right = (const float*)d_in[1];
  const float* w1 = (const float*)d_in[6];
  const float* b1 = (const float*)d_in[7];
  const float* g1 = (const float*)d_in[8];
  const float* be1 = (const float*)d_in[9];
  const float* m1 = (const float*)d_in[10];
  const float* v1 = (const float*)d_in[11];
  const float* p1 = (const float*)d_in[12];
  const float* w2 = (const float*)d_in[13];
  const float* b2 = (const float*)d_in[14];
  const float* g2 = (const float*)d_in[15];
  const float* be2 = (const float*)d_in[16];
  const float* m2 = (const float*)d_in[17];
  const float* v2 = (const float*)d_in[18];
  const float* p2 = (const float*)d_in[19];
  const float* wq = (const float*)d_in[20];
  const float* wk = (const float*)d_in[22];

  float* c0   = (float*)(ws + O_C0);
  float* c1   = (float*)(ws + O_C1);
  float* feaL = (float*)(ws + O_FEAL);
  float* feaR = (float*)(ws + O_FEAR);
  float* fl1  = (float*)(ws + O_FL1);
  float* fr1  = (float*)(ws + O_FR1);
  _Float16* fl1p = (_Float16*)(ws + O_FL1P);
  _Float16* fr1p = (_Float16*)(ws + O_FR1P);
  float* ybuf = (float*)(ws + O_YB);
  float* g1o  = (float*)(ws + O_G1);
  float* g2o  = (float*)(ws + O_G2);
  _Float16* col = (_Float16*)(ws + O_COL);
  float* qraw = (float*)(ws + O_QRAW);
  _Float16* Qp = (_Float16*)(ws + O_QP);
  _Float16* Kp = (_Float16*)(ws + O_KP);
  _Float16* w1h = (_Float16*)(ws + O_W1H);
  _Float16* w2h = (_Float16*)(ws + O_W2H);
  _Float16* wqh = (_Float16*)(ws + O_WQH);
  _Float16* wkh = (_Float16*)(ws + O_WKH);
  _Float16* M16 = (_Float16*)(ws + O_M16);
  _Float16* xt16 = (_Float16*)(ws + O_XT16);
  float* xT = (float*)(ws + O_XT);

  const int NCHW = BB * CC * HW;        // 8388608

  // seed cost volumes and running features
  hipMemcpyAsync(c0, d_in[4], 134217728ull, hipMemcpyDeviceToDevice, stream);
  hipMemcpyAsync(c1, d_in[5], 134217728ull, hipMemcpyDeviceToDevice, stream);
  hipMemcpyAsync(feaL, d_in[2], (size_t)NCHW * 4, hipMemcpyDeviceToDevice, stream);
  hipMemcpyAsync(feaR, d_in[3], (size_t)NCHW * 4, hipMemcpyDeviceToDevice, stream);

  // pack weights f32 -> f16 (OIHW flat == Bt[N x K] with k = c*9+r*3+s)
  k_f32_to_f16<<<gs(73728), 256, 0, stream>>>(w1, w1h, 73728);
  k_f32_to_f16<<<gs(73728), 256, 0, stream>>>(w2, w2h, 73728);
  k_f32_to_f16<<<gs(16384), 256, 0, stream>>>(wq, wqh, 16384);
  k_f32_to_f16<<<gs(16384), 256, 0, stream>>>(wk, wkh, 16384);

  for (int i = 0; i < 4; ++i) {
    const _Float16* w1i = w1h + (size_t)i * CH * CC * 9;  // (32 x 576)
    const _Float16* w2i = w2h + (size_t)i * CC * CH * 9;  // (64 x 288)

    for (int side = 0; side < 2; ++side) {
      float* fea = side ? feaR : feaL;
      float* f1  = side ? fr1 : fl1;
      _Float16* f1p = side ? fr1p : fl1p;
      for (int b = 0; b < BB; ++b) {
        const float* xin = fea + (size_t)b * CC * HW;
        // conv1: 64 -> 32, K = 576
        k_im2col<<<gs((long long)HW * 576), 256, 0, stream>>>(xin, col, CC, HW * 576);
        launch_gemm(stream, col, w1i, g1o, HW, CH, 576, 1, 0, 0, 0, false);
        k_bn_prelu<<<gs((long long)CH * HW), 256, 0, stream>>>(
            g1o, ybuf, (_Float16*)nullptr,
            b1 + i * CH, g1 + i * CH, be1 + i * CH, m1 + i * CH, v1 + i * CH,
            p1 + i, CH, CH * HW);
        // conv2: 32 -> 64, K = 288
        k_im2col<<<gs((long long)HW * 288), 256, 0, stream>>>(ybuf, col, CH, HW * 288);
        launch_gemm(stream, col, w2i, g2o, HW, CC, 288, 1, 0, 0, 0, false);
        k_bn_prelu<<<gs((long long)CC * HW), 256, 0, stream>>>(
            g2o, f1 + (size_t)b * CC * HW, f1p + (size_t)b * HW * CC,
            b2 + i * CC, g2 + i * CC, be2 + i * CC, m2 + i * CC, v2 + i * CC,
            p2 + i, CC, CC * HW);
      }
    }

    // Q = 1x1 conv(fl1), K = 1x1 conv(fr1); mean over W cancels the bias
    launch_gemm(stream, fl1p, wqh + (size_t)i * 4096, qraw,
                BB * HW, CC, CC, 1, 0, 0, 0, false);
    k_qk_post<<<gs(BH * CC), 256, 0, stream>>>(qraw, Qp);
    launch_gemm(stream, fr1p, wkh + (size_t)i * 4096, qraw,
                BB * HW, CC, CC, 1, 0, 0, 0, false);
    k_qk_post<<<gs(BH * CC), 256, 0, stream>>>(qraw, Kp);

    // score = Qt*K per (b,h); c0 += score, c1 += score^T = Kt*Q
    launch_gemm(stream, Qp, Kp, c0, WW, WW, CC, BH,
                (long long)WW * CC, (long long)WW * CC, (long long)WW * WW, true);
    launch_gemm(stream, Kp, Qp, c1, WW, WW, CC, BH,
                (long long)WW * CC, (long long)WW * CC, (long long)WW * WW, true);

    // fea += fea1
    k_add<<<gs(NCHW), 256, 0, stream>>>(feaL, fl1, NCHW);
    k_add<<<gs(NCHW), 256, 0, stream>>>(feaR, fr1, NCHW);
  }

  // softmax + disparity
  k_softmax_disp<<<BH * WW, 256, 0, stream>>>(c0, out + OUT_M0, out + OUT_D1);
  k_softmax_disp<<<BH * WW, 256, 0, stream>>>(c1, out + OUT_M1, out + OUT_D2);

  // validity maps
  k_relax_V<<<BH * WW, 256, 0, stream>>>(out + OUT_M0, out + OUT_M1, out + OUT_VL);
  k_relax_V<<<BH * WW, 256, 0, stream>>>(out + OUT_M1, out + OUT_M0, out + OUT_VR);

  // out_left: x_leftT = M_r2l @ x_right  (batched 256x256 @ 256x64)
  k_f32_to_f16<<<gs(33554432), 256, 0, stream>>>(out + OUT_M0, M16, 33554432);
  k_xT16<<<gs(NCHW), 256, 0, stream>>>(x_right, xt16, NCHW);
  launch_gemm(stream, M16, xt16, xT, WW, CC, WW, BH,
              (long long)WW * WW, (long long)CC * WW, (long long)WW * CC, false);
  k_blend<<<gs(NCHW), 256, 0, stream>>>(x_left, xT, out + OUT_VL, out + OUT_L, NCHW);

  // out_right: x_rightT = M_l2r @ x_left
  k_f32_to_f16<<<gs(33554432), 256, 0, stream>>>(out + OUT_M1, M16, 33554432);
  k_xT16<<<gs(NCHW), 256, 0, stream>>>(x_left, xt16, NCHW);
  launch_gemm(stream, M16, xt16, xT, WW, CC, WW, BH,
              (long long)WW * WW, (long long)CC * WW, (long long)WW * CC, false);
  k_blend<<<gs(NCHW), 256, 0, stream>>>(x_right, xT, out + OUT_VR, out + OUT_R, NCHW);
}
